// GAT_57921928954041
// MI455X (gfx1250) — matrix operand order repaired and checked by reading, not compiled, not run on hardware
//
#include <hip/hip_runtime.h>
#include <hip/hip_bf16.h>

// ---------------------------------------------------------------------------
// GAT forward (2 layers + classifier) for MI455X / gfx1250.
// GEMMs use v_wmma_f32_16x16x32_f16 (f16 inputs, f32 accumulation).
// Edge softmax/aggregation uses native f32/u32 global atomics (memory-bound
// part: ~870MB gather + ~870MB scatter RMW per layer at 23.3 TB/s).
// Workspace layout (needs ~120 MB):
//   bufH, bufA : N*128 f32 each   (ping-pong activations / accumulators)
//   al, ar, m, denom : N*8 f32 each
//   w1s, w2s, wcs : f16 swizzled weights (B-fragment layout)
// ---------------------------------------------------------------------------

typedef __attribute__((ext_vector_type(16))) _Float16 v16h;
typedef __attribute__((ext_vector_type(8)))  float    v8f;

#define KDIM   128
#define KSTEPS 4        // 128 / 32
#define HEADS  8
#define CH     16
#define FDIM   128
#define SLOPE  0.2f

// ---------------- weight pre-swizzle into WMMA B-fragment layout -----------
// Layout: Wswz[((nb*KSTEPS + ks)*32 + lane)*16 + e]
//   column n = nb*16 + (lane&15); K = ks*32 + (lane>>4)*16 + e
__global__ void gat_swizzle_w(const float* __restrict__ W,
                              _Float16* __restrict__ Wswz,
                              int Nout, int NB)
{
    int t = blockIdx.x * blockDim.x + threadIdx.x;
    if (t >= NB * KSTEPS * 32) return;
    int lane = t & 31;
    int ks   = (t >> 5) & (KSTEPS - 1);
    int half = lane >> 4;
    int nsub = lane & 15;
    int nb   = t >> 7;
    int n    = nb * 16 + nsub;
    _Float16* op = Wswz + (size_t)t * 16;
#pragma unroll
    for (int e = 0; e < 16; ++e) {
        int k = ks * 32 + half * 16 + e;
        float v = (n < Nout) ? W[(size_t)k * Nout + n] : 0.0f;
        op[e] = (_Float16)v;
    }
}

// ---------------- WMMA GEMM: Y[N x Ncols] = X[N x 128] @ W (+bias) ---------
// One workgroup = 16 rows; wave w computes column tile [16w, 16w+16).
__global__ void __launch_bounds__(256)
gat_gemm_wmma(const float* __restrict__ X,
              const _Float16* __restrict__ Wswz,
              const float* __restrict__ bias,
              float* __restrict__ Y,
              int Nrows, int NB, int Ncols)
{
    __shared__ _Float16 xs[16 * KDIM];
    const int row0 = blockIdx.x << 4;

    // stage 16x128 f32 tile -> f16 LDS (4 KB)
    for (int i = threadIdx.x; i < 16 * KDIM; i += blockDim.x) {
        int r = i >> 7, k = i & (KDIM - 1);
        int rr = row0 + r;
        if (rr >= Nrows) rr = Nrows - 1;
        xs[i] = (_Float16)X[(size_t)rr * KDIM + k];
    }
    __syncthreads();

    const int wave = threadIdx.x >> 5;
    const int lane = threadIdx.x & 31;
    if (wave >= NB) return;               // wave-uniform exit: EXEC stays full
    const int half = lane >> 4;
    const int nsub = lane & 15;

    v8f acc = {};
    const _Float16* wp_base = Wswz + ((size_t)wave * KSTEPS * 32 + lane) * 16;
#pragma unroll
    for (int ks = 0; ks < KSTEPS; ++ks) {
        // A fragment: row = nsub; elems 0..15 -> K = ks*32 + half*16 + e
        v16h a, b;
        const _Float16* ap = &xs[nsub * KDIM + ks * 32 + half * 16];
#pragma unroll
        for (int e = 0; e < 16; ++e) { a[e] = ap[e]; }
        // B fragment: contiguous 32B per lane from the swizzled weights
        const _Float16* bp = wp_base + (size_t)ks * 32 * 16;
        if (ks + 1 < KSTEPS)
            __builtin_prefetch(wp_base + (size_t)(ks + 1) * 32 * 16, 0, 1);
#pragma unroll
        for (int e = 0; e < 16; ++e) b[e] = bp[e];
        acc = __builtin_amdgcn_wmma_f32_16x16x32_f16(
            /*neg_a=*/false, a, /*neg_b=*/false, b,
            /*c_mod=*/(short)0, acc, /*reuse_a=*/false, /*reuse_b=*/false);
    }

    // D layout: col = nsub (+tile), row = v + 8*half
    const int n = wave * 16 + nsub;
    if (n < Ncols) {
        const float bv = bias ? bias[n] : 0.0f;
#pragma unroll
        for (int v = 0; v < 8; ++v) {
            int r = row0 + v + half * 8;
            if (r < Nrows) Y[(size_t)r * Ncols + n] = acc[v] + bv;
        }
    }
}

// ---------------- per-node attention coefficients --------------------------
__global__ void gat_alphas(const float* __restrict__ Hm,
                           const float* __restrict__ att_l,
                           const float* __restrict__ att_r,
                           float* __restrict__ al, float* __restrict__ ar,
                           int Nn)
{
    int t = blockIdx.x * blockDim.x + threadIdx.x;
    if (t >= Nn * HEADS) return;
    int n = t >> 3, h = t & 7;
    const float* hp = Hm + (size_t)n * FDIM + h * CH;
    const float* lp = att_l + h * CH;
    const float* rp = att_r + h * CH;
    float sl = 0.f, sr = 0.f;
#pragma unroll
    for (int c = 0; c < CH; ++c) { float v = hp[c]; sl += v * lp[c]; sr += v * rp[c]; }
    al[t] = sl; ar[t] = sr;
}

// ---------------- fill ------------------------------------------------------
__global__ void gat_fill(float* __restrict__ p, float v, long n)
{
    long i = (long)blockIdx.x * blockDim.x + threadIdx.x;
    long s = (long)gridDim.x * blockDim.x;
    for (; i < n; i += s) p[i] = v;
}

// ---------------- order-preserving float <-> u32 for atomicMax -------------
__device__ __forceinline__ unsigned enc_f(float f) {
    unsigned u = __float_as_uint(f);
    return (u & 0x80000000u) ? ~u : (u | 0x80000000u);
}
__device__ __forceinline__ float dec_f(unsigned e) {
    unsigned u = (e & 0x80000000u) ? (e & 0x7FFFFFFFu) : ~e;
    return __uint_as_float(u);
}

// ---------------- edge pass 1: segment max (all edges, incl. masked) -------
__global__ void gat_edge_max(const int* __restrict__ ei, int E0, int Etot,
                             const float* __restrict__ al,
                             const float* __restrict__ ar,
                             unsigned* __restrict__ menc)
{
    int e = blockIdx.x * blockDim.x + threadIdx.x;
    if (e >= Etot) return;
    int s, d;
    if (e < E0) { s = ei[e]; d = ei[E0 + e]; }
    else        { s = d = e - E0; }
#pragma unroll
    for (int h = 0; h < HEADS; ++h) {
        float sc = al[s * HEADS + h] + ar[d * HEADS + h];
        sc = sc > 0.f ? sc : SLOPE * sc;
        atomicMax(&menc[d * HEADS + h], enc_f(sc));
    }
}

__global__ void gat_decode_max(float* __restrict__ m, long n)
{
    long i = (long)blockIdx.x * blockDim.x + threadIdx.x;
    if (i >= n) return;
    m[i] = dec_f(__float_as_uint(m[i]));
}

// ---------------- edge pass 2: sum of exp (masked edges skipped) -----------
__global__ void gat_edge_expsum(const int* __restrict__ ei, int E0, int Etot,
                                const float* __restrict__ al,
                                const float* __restrict__ ar,
                                const float* __restrict__ m,
                                float* __restrict__ denom)
{
    int e = blockIdx.x * blockDim.x + threadIdx.x;
    if (e >= Etot) return;
    int s, d;
    if (e < E0) { s = ei[e]; d = ei[E0 + e]; if (s == d) return; }
    else        { s = d = e - E0; }
#pragma unroll
    for (int h = 0; h < HEADS; ++h) {
        float sc = al[s * HEADS + h] + ar[d * HEADS + h];
        sc = sc > 0.f ? sc : SLOPE * sc;
        atomicAdd(&denom[d * HEADS + h], expf(sc - m[d * HEADS + h]));
    }
}

// ---------------- edge pass 3: weighted scatter (1 thread / edge,head) -----
__global__ void gat_edge_scatter(const int* __restrict__ ei, int E0, int Etot,
                                 const float* __restrict__ al,
                                 const float* __restrict__ ar,
                                 const float* __restrict__ m,
                                 const float* __restrict__ denom,
                                 const float* __restrict__ Hsrc,
                                 float* __restrict__ Agg)
{
    int t = blockIdx.x * blockDim.x + threadIdx.x;
    if (t >= Etot * HEADS) return;
    int e = t >> 3, h = t & 7;
    int s, d;
    if (e < E0) { s = ei[e]; d = ei[E0 + e]; if (s == d) return; }
    else        { s = d = e - E0; }
    float sc = al[s * HEADS + h] + ar[d * HEADS + h];
    sc = sc > 0.f ? sc : SLOPE * sc;
    float alpha = expf(sc - m[d * HEADS + h]) /
                  fmaxf(denom[d * HEADS + h], 1e-16f);
    const float* hp = Hsrc + (size_t)s * FDIM + h * CH;
    float*       op = Agg  + (size_t)d * FDIM + h * CH;
    float4 h0 = *(const float4*)(hp + 0);
    float4 h1 = *(const float4*)(hp + 4);
    float4 h2 = *(const float4*)(hp + 8);
    float4 h3 = *(const float4*)(hp + 12);
    atomicAdd(op + 0,  h0.x * alpha); atomicAdd(op + 1,  h0.y * alpha);
    atomicAdd(op + 2,  h0.z * alpha); atomicAdd(op + 3,  h0.w * alpha);
    atomicAdd(op + 4,  h1.x * alpha); atomicAdd(op + 5,  h1.y * alpha);
    atomicAdd(op + 6,  h1.z * alpha); atomicAdd(op + 7,  h1.w * alpha);
    atomicAdd(op + 8,  h2.x * alpha); atomicAdd(op + 9,  h2.y * alpha);
    atomicAdd(op + 10, h2.z * alpha); atomicAdd(op + 11, h2.w * alpha);
    atomicAdd(op + 12, h3.x * alpha); atomicAdd(op + 13, h3.y * alpha);
    atomicAdd(op + 14, h3.z * alpha); atomicAdd(op + 15, h3.w * alpha);
}

// ---------------- bias + ELU ------------------------------------------------
__global__ void gat_bias_elu(const float* __restrict__ in,
                             const float* __restrict__ bias,
                             float* __restrict__ out, long n)
{
    long t = (long)blockIdx.x * blockDim.x + threadIdx.x;
    if (t >= n) return;
    float v = in[t] + bias[t & (FDIM - 1)];
    out[t] = v > 0.f ? v : (expf(v) - 1.0f);
}

// ---------------------------------------------------------------------------
static inline int cdiv(long a, long b) { return (int)((a + b - 1) / b); }

extern "C" void kernel_launch(void* const* d_in, const int* in_sizes, int n_in,
                              void* d_out, int out_size, void* d_ws, size_t ws_size,
                              hipStream_t stream)
{
    const float* x     = (const float*)d_in[0];
    const int*   ei    = (const int*)d_in[1];   // [2, E0]
    // d_in[2] edge_weight: unused (matches reference)
    const float* W1    = (const float*)d_in[3];
    const float* attl1 = (const float*)d_in[4];
    const float* attr1 = (const float*)d_in[5];
    const float* b1    = (const float*)d_in[6];
    const float* W2    = (const float*)d_in[7];
    const float* attl2 = (const float*)d_in[8];
    const float* attr2 = (const float*)d_in[9];
    const float* b2    = (const float*)d_in[10];
    const float* Wc    = (const float*)d_in[11];
    const float* bc    = (const float*)d_in[12];
    float* out = (float*)d_out;

    const int Nn   = in_sizes[0] / KDIM;
    const int E0   = in_sizes[1] / 2;
    const int Etot = E0 + Nn;
    const int NCLS = 40;

    // ---- workspace carve-up ----
    char* ws = (char*)d_ws;
    size_t off = 0;
    auto carve = [&](size_t bytes) -> void* {
        void* p = ws + off;
        off += (bytes + 255) & ~(size_t)255;
        return p;
    };
    float* bufH  = (float*)carve((size_t)Nn * FDIM * 4);
    float* bufA  = (float*)carve((size_t)Nn * FDIM * 4);
    float* al    = (float*)carve((size_t)Nn * HEADS * 4);
    float* ar    = (float*)carve((size_t)Nn * HEADS * 4);
    float* m     = (float*)carve((size_t)Nn * HEADS * 4);
    float* denom = (float*)carve((size_t)Nn * HEADS * 4);
    _Float16* w1s = (_Float16*)carve((size_t)8 * KSTEPS * 32 * 16 * 2);
    _Float16* w2s = (_Float16*)carve((size_t)8 * KSTEPS * 32 * 16 * 2);
    _Float16* wcs = (_Float16*)carve((size_t)3 * KSTEPS * 32 * 16 * 2);

    const long NF = (long)Nn * FDIM;
    const long NH = (long)Nn * HEADS;
    const int gRows  = cdiv(Nn, 16);
    const int gNH    = cdiv(NH, 256);
    const int gNF    = cdiv(NF, 256);
    const int gEdge  = cdiv(Etot, 256);
    const int gEdgeH = cdiv((long)Etot * HEADS, 256);

    // ---- weight swizzles (f32 -> f16 B-fragment layout) ----
    gat_swizzle_w<<<4, 256, 0, stream>>>(W1, w1s, FDIM, 8);
    gat_swizzle_w<<<4, 256, 0, stream>>>(W2, w2s, FDIM, 8);
    gat_swizzle_w<<<2, 256, 0, stream>>>(Wc, wcs, NCLS, 3);

    // ================= layer 1 =================
    gat_gemm_wmma<<<gRows, 256, 0, stream>>>(x, w1s, nullptr, bufH, Nn, 8, FDIM);
    gat_alphas<<<gNH, 256, 0, stream>>>(bufH, attl1, attr1, al, ar, Nn);
    gat_fill<<<1024, 256, 0, stream>>>(m, 0.0f, NH);        // 0u == encoded -inf floor
    gat_fill<<<1024, 256, 0, stream>>>(denom, 0.0f, NH);
    gat_fill<<<2048, 256, 0, stream>>>(bufA, 0.0f, NF);
    gat_edge_max<<<gEdge, 256, 0, stream>>>(ei, E0, Etot, al, ar, (unsigned*)m);
    gat_decode_max<<<gNH, 256, 0, stream>>>(m, NH);
    gat_edge_expsum<<<gEdge, 256, 0, stream>>>(ei, E0, Etot, al, ar, m, denom);
    gat_edge_scatter<<<gEdgeH, 256, 0, stream>>>(ei, E0, Etot, al, ar, m, denom,
                                                 bufH, bufA);
    gat_bias_elu<<<gNF, 256, 0, stream>>>(bufA, b1, bufH, NF);  // act1 -> bufH

    // ================= layer 2 =================
    gat_gemm_wmma<<<gRows, 256, 0, stream>>>(bufH, w2s, nullptr, bufA, Nn, 8, FDIM);
    gat_alphas<<<gNH, 256, 0, stream>>>(bufA, attl2, attr2, al, ar, Nn);
    gat_fill<<<1024, 256, 0, stream>>>(m, 0.0f, NH);
    gat_fill<<<1024, 256, 0, stream>>>(denom, 0.0f, NH);
    gat_fill<<<2048, 256, 0, stream>>>(bufH, 0.0f, NF);
    gat_edge_max<<<gEdge, 256, 0, stream>>>(ei, E0, Etot, al, ar, (unsigned*)m);
    gat_decode_max<<<gNH, 256, 0, stream>>>(m, NH);
    gat_edge_expsum<<<gEdge, 256, 0, stream>>>(ei, E0, Etot, al, ar, m, denom);
    gat_edge_scatter<<<gEdgeH, 256, 0, stream>>>(ei, E0, Etot, al, ar, m, denom,
                                                 bufA, bufH);
    gat_bias_elu<<<gNF, 256, 0, stream>>>(bufH, b2, bufA, NF);  // act2 -> bufA

    // ================= classifier (fused bias) =================
    gat_gemm_wmma<<<gRows, 96, 0, stream>>>(bufA, wcs, bc, out, Nn, 3, NCLS);
}
